// LSTM_17575006175928
// MI455X (gfx1250) — compile-verified, matrix operand order checked by reading, not run on hardware
//
#include <hip/hip_runtime.h>

// ---------------------------------------------------------------------------
// MI455X (gfx1250) LSTM inference.
//   - bf16 WMMA (v_wmma_f32_16x16x32_bf16) everywhere, fp32 accumulate.
//   - Per-layer: (1) big parallel WMMA GEMM  G = X @ Wih^T + (b_ih+b_hh),
//                   stored gate-interleaved so the recurrence reads b128 quads
//               (2) persistent recurrence kernel: Whh held ENTIRELY IN VGPRs
//                   (16 B-fragments = 128 VGPRs per wave, loop-invariant),
//                   h tile in LDS (4KB), c in registers, 512 serial steps.
//   - MLP head: two WMMA GEMMs (+ReLU), final N=1 projection as a dot kernel.
// ---------------------------------------------------------------------------

typedef __attribute__((ext_vector_type(16))) __bf16 v16bf;
typedef __attribute__((ext_vector_type(8)))  float  v8f;

union Frag {            // one WMMA 16-bit A/B operand fragment per lane
  int4  q[2];           // two 16-byte loads
  int   i[8];
  v16bf v;
};

__device__ __forceinline__ unsigned short f2bf(float f) {
  unsigned int u = __float_as_uint(f);
  u += 0x7FFFu + ((u >> 16) & 1u);          // round-to-nearest-even
  return (unsigned short)(u >> 16);
}
__device__ __forceinline__ float bf2f(unsigned short h) {
  return __uint_as_float(((unsigned int)h) << 16);
}

// ---------------------------------------------------------------------------
// fp32 [rows,K] -> bf16 [rows,KP] with zero padding (weights + activations)
// ---------------------------------------------------------------------------
__global__ void pad_convert_bf16(const float* __restrict__ src,
                                 unsigned short* __restrict__ dst,
                                 int rows, int K, int KP) {
  long idx = (long)blockIdx.x * blockDim.x + threadIdx.x;
  long tot = (long)rows * KP;
  if (idx >= tot) return;
  int  k = (int)(idx % KP);
  long r = idx / KP;
  float v = (k < K) ? src[r * (long)K + k] : 0.0f;
  dst[idx] = f2bf(v);
}

__global__ void bias_combine(const float* __restrict__ a,
                             const float* __restrict__ b,
                             float* __restrict__ out, int n) {
  int i = blockIdx.x * blockDim.x + threadIdx.x;
  if (i < n) out[i] = a[i] + b[i];
}

// ---------------------------------------------------------------------------
// Generic WMMA GEMM:  out[M,N] = A[M,KP](bf16) @ W[N,KP]^T(bf16) + bias[N]
// One wave computes a 16x64 tile (4 accumulators), K-loop of 16x16x32 WMMAs.
// PERM_GATES (N==512 only): store fp32 at  mrow*512 + (n%128)*4 + n/128
// so the LSTM reads each lane's {i,f,g,o} quad with one global_load_b128.
// ---------------------------------------------------------------------------
template <int KP, bool RELU, bool OUT_BF16, bool PERM_GATES>
__global__ __launch_bounds__(256) void gemm_wmma_bf16(
    const unsigned short* __restrict__ A,   // bf16 [M, KP]
    const unsigned short* __restrict__ W,   // bf16 [N, KP]
    const float* __restrict__ bias,         // [N]
    void* __restrict__ out, int M, int N) {
  const int lane = threadIdx.x & 31;
  const int wid  = (int)((blockIdx.x * (long)blockDim.x + threadIdx.x) >> 5);
  const int nt64 = N >> 6;
  const int mtile = wid / nt64;
  const int n0 = (wid - mtile * nt64) << 6;
  if (mtile * 16 >= M) return;             // wave-uniform: EXEC stays all-ones
  const int half = lane >> 4;
  const int ml   = lane & 15;
  const v8f zero = {0.f,0.f,0.f,0.f,0.f,0.f,0.f,0.f};
  v8f acc[4] = {zero, zero, zero, zero};
  const long arow = (long)(mtile * 16 + ml) * KP;
#pragma unroll
  for (int kt = 0; kt < KP / 32; ++kt) {
    Frag af;
    const int4* pa = (const int4*)(A + arow + kt * 32 + half * 8);
    af.q[0] = pa[0];
    af.q[1] = pa[2];                        // +16 bf16 elements
#pragma unroll
    for (int g = 0; g < 4; ++g) {
      Frag bf;
      const int4* pb =
          (const int4*)(W + (long)(n0 + g * 16 + ml) * KP + kt * 32 + half * 8);
      bf.q[0] = pb[0];
      bf.q[1] = pb[2];
      acc[g] = __builtin_amdgcn_wmma_f32_16x16x32_bf16(
          false, af.v, false, bf.v, (short)0, acc[g], false, false);
    }
  }
#pragma unroll
  for (int g = 0; g < 4; ++g) {
    int n = n0 + g * 16 + ml;
    float bv = bias[n];
#pragma unroll
    for (int r = 0; r < 8; ++r) {
      int mrow = mtile * 16 + half * 8 + r;   // C layout: M = r + (lane/16)*8
      float v = acc[g][r] + bv;
      if (RELU) v = v > 0.f ? v : 0.f;
      if (OUT_BF16) {
        ((unsigned short*)out)[(long)mrow * N + n] = f2bf(v);
      } else if (PERM_GATES) {
        ((float*)out)[(long)mrow * N + (n & 127) * 4 + (n >> 7)] = v;
      } else {
        ((float*)out)[(long)mrow * N + n] = v;
      }
    }
  }
}

// ---------------------------------------------------------------------------
// Persistent LSTM recurrence kernel, one layer.
// Grid = B/16 workgroups, 256 threads (8 waves). Wave w owns h columns
// [w*16, w*16+16): its 4 gate tiles are ntile = w + g*8 (n = w*16 + g*128).
// All 16 Whh B-fragments (128 VGPRs) loaded once from global and held in
// registers for all T steps. h tile (16x128 bf16) in LDS. gbuf holds the
// precomputed x-projections + biases, gate-interleaved: [b][t][col][gate].
// ---------------------------------------------------------------------------
__global__ __launch_bounds__(256) void lstm_layer_kernel(
    const float* __restrict__ gbuf,          // [B, T, 128, 4] (col, i/f/g/o)
    const unsigned short* __restrict__ whh,  // bf16 [512, 128] (row n, col k)
    unsigned short* __restrict__ yout,       // bf16 [B, T, 128]
    int Tn) {
  __shared__ alignas(16) unsigned short h_lds[16 * 128];  // 4 KB
  const int lane = threadIdx.x & 31;
  const int wave = threadIdx.x >> 5;
  const int b0   = blockIdx.x * 16;
  const int half = lane >> 4;
  const int ml   = lane & 15;
  const int ncol = wave * 16 + ml;              // this lane's h / gate column

  // Load all Whh B-fragments into registers (loop-invariant across T).
  // B[k][n] = Whh[n][k]; 16-bit 32x16 B layout: n = lane%16 striped on lanes,
  // K pairs at kt*32 + (lane/16)*8 (+16) -> two 16B global loads per fragment.
  Frag bw[4][4];                                // 128 VGPRs
#pragma unroll
  for (int kt = 0; kt < 4; ++kt) {
#pragma unroll
    for (int g = 0; g < 4; ++g) {
      int n = (g * 8 + wave) * 16 + ml;
      const int4* pb = (const int4*)(whh + (long)n * 128 + kt * 32 + half * 8);
      bw[kt][g].q[0] = pb[0];
      bw[kt][g].q[1] = pb[2];
    }
  }

  for (int i = threadIdx.x; i < 16 * 128; i += 256) h_lds[i] = 0;  // h0 = 0
  __syncthreads();

  float cst[8];                                 // c0 = 0
#pragma unroll
  for (int r = 0; r < 8; ++r) cst[r] = 0.f;
  const v8f zero = {0.f,0.f,0.f,0.f,0.f,0.f,0.f,0.f};

  for (int t = 0; t < Tn; ++t) {
    // Prefetch this step's gate preactivations (independent of h) so the
    // global loads overlap the WMMA chain. One b128 quad per row.
    float4 gp[8];
#pragma unroll
    for (int r = 0; r < 8; ++r) {
      int m = half * 8 + r;
      gp[r] = *(const float4*)(gbuf + ((long)(b0 + m) * Tn + t) * 512 + ncol * 4);
    }

    v8f acc[4] = {zero, zero, zero, zero};      // i, f, g, o tiles
#pragma unroll
    for (int kt = 0; kt < 4; ++kt) {
      Frag af;                                  // h tile as WMMA A fragment
      const int4* pa = (const int4*)(&h_lds[ml * 128 + kt * 32 + half * 8]);
      af.q[0] = pa[0];
      af.q[1] = pa[2];
#pragma unroll
      for (int g = 0; g < 4; ++g) {
        acc[g] = __builtin_amdgcn_wmma_f32_16x16x32_bf16(
            false, af.v, false, bw[kt][g].v, (short)0, acc[g], false, false);
      }
    }
    __syncthreads();   // all waves done reading h_lds before overwrite
#pragma unroll
    for (int r = 0; r < 8; ++r) {
      int  m  = half * 8 + r;                   // row within 16-batch tile
      float gi = acc[0][r] + gp[r].x;
      float gf = acc[1][r] + gp[r].y;
      float gg = acc[2][r] + gp[r].z;
      float go = acc[3][r] + gp[r].w;
      float i_ = 1.f / (1.f + __expf(-gi));
      float f_ = 1.f / (1.f + __expf(-gf));
      float g_ = tanhf(gg);
      float o_ = 1.f / (1.f + __expf(-go));
      float c  = f_ * cst[r] + i_ * g_;
      cst[r]   = c;
      float h  = o_ * tanhf(c);
      unsigned short hb = f2bf(h);
      h_lds[m * 128 + ncol] = hb;
      yout[((long)(b0 + m) * Tn + t) * 128 + ncol] = hb;
    }
    __syncthreads();   // new h visible to all waves
  }
}

// ---------------------------------------------------------------------------
// Build MLP input: [x(40) | y2(128) | pad(24)] as bf16 [M, 192]
// ---------------------------------------------------------------------------
__global__ void concat_kernel(const float* __restrict__ x,
                              const unsigned short* __restrict__ y2,
                              unsigned short* __restrict__ dst, long M) {
  long idx = (long)blockIdx.x * blockDim.x + threadIdx.x;
  long tot = M * 192;
  if (idx >= tot) return;
  int  k = (int)(idx % 192);
  long m = idx / 192;
  unsigned short v;
  if (k < 40)       v = f2bf(x[m * 40 + k]);
  else if (k < 168) v = y2[m * 128 + (k - 40)];
  else              v = 0;
  dst[idx] = v;
}

// ---------------------------------------------------------------------------
// Final N=1 projection: out[m] = h2[m,:] . W3 + b3
// ---------------------------------------------------------------------------
__global__ void final_proj(const unsigned short* __restrict__ h2,
                           const unsigned short* __restrict__ w3,
                           const float* __restrict__ b3,
                           float* __restrict__ out, long M) {
  long m = (long)blockIdx.x * blockDim.x + threadIdx.x;
  if (m >= M) return;
  float s = b3[0];
  const unsigned short* row = h2 + m * 128;
#pragma unroll 8
  for (int k = 0; k < 128; ++k) s += bf2f(row[k]) * bf2f(w3[k]);
  out[m] = s;
}

// ---------------------------------------------------------------------------
extern "C" void kernel_launch(void* const* d_in, const int* in_sizes, int n_in,
                              void* d_out, int out_size, void* d_ws, size_t ws_size,
                              hipStream_t stream) {
  const float* x      = (const float*)d_in[0];
  const float* Wih[3] = {(const float*)d_in[3],  (const float*)d_in[7],  (const float*)d_in[11]};
  const float* Whh[3] = {(const float*)d_in[4],  (const float*)d_in[8],  (const float*)d_in[12]};
  const float* bih[3] = {(const float*)d_in[5],  (const float*)d_in[9],  (const float*)d_in[13]};
  const float* bhh[3] = {(const float*)d_in[6],  (const float*)d_in[10], (const float*)d_in[14]};
  const float* W1 = (const float*)d_in[15]; const float* b1 = (const float*)d_in[16];
  const float* W2 = (const float*)d_in[17]; const float* b2 = (const float*)d_in[18];
  const float* W3 = (const float*)d_in[19]; const float* b3 = (const float*)d_in[20];

  const long M  = 256L * 512L;   // B*T = 131072
  const int  Tn = 512;

  // workspace carving (~470 MB)
  char*  ws  = (char*)d_ws;
  size_t off = 0;
  auto carve = [&](size_t bytes) {
    char* p = ws + off;
    off = (off + bytes + 255) & ~(size_t)255;
    return p;
  };
  float*          gbuf  = (float*)         carve(M * 512 * 4);
  unsigned short* yA    = (unsigned short*)carve(M * 128 * 2);
  unsigned short* yB    = (unsigned short*)carve(M * 128 * 2);
  unsigned short* xbf   = (unsigned short*)carve(M * 64 * 2);
  unsigned short* mlpin = (unsigned short*)carve(M * 192 * 2);
  unsigned short* h1    = (unsigned short*)carve(M * 128 * 2);
  unsigned short* h2    = (unsigned short*)carve(M * 128 * 2);
  unsigned short* wihb0 = (unsigned short*)carve(512 * 64 * 2);
  unsigned short* wihb1 = (unsigned short*)carve(512 * 128 * 2);
  unsigned short* wihb2 = (unsigned short*)carve(512 * 128 * 2);
  unsigned short* whhb[3];
  for (int l = 0; l < 3; ++l) whhb[l] = (unsigned short*)carve(512 * 128 * 2);
  float* biasc[3];
  for (int l = 0; l < 3; ++l) biasc[l] = (float*)carve(512 * 4);
  unsigned short* w1b = (unsigned short*)carve(128 * 192 * 2);
  unsigned short* w2b = (unsigned short*)carve(128 * 128 * 2);
  unsigned short* w3b = (unsigned short*)carve(128 * 2);

  auto blocks = [](long tot) { return (unsigned)((tot + 255) / 256); };

  // --- conversions / packing ---
  pad_convert_bf16<<<blocks(M * 64), 256, 0, stream>>>(x, xbf, (int)M, 40, 64);
  pad_convert_bf16<<<blocks(512 * 64), 256, 0, stream>>>(Wih[0], wihb0, 512, 40, 64);
  pad_convert_bf16<<<blocks(512 * 128), 256, 0, stream>>>(Wih[1], wihb1, 512, 128, 128);
  pad_convert_bf16<<<blocks(512 * 128), 256, 0, stream>>>(Wih[2], wihb2, 512, 128, 128);
  for (int l = 0; l < 3; ++l) {
    pad_convert_bf16<<<blocks(512 * 128), 256, 0, stream>>>(Whh[l], whhb[l], 512, 128, 128);
    bias_combine<<<2, 256, 0, stream>>>(bih[l], bhh[l], biasc[l], 512);
  }
  pad_convert_bf16<<<blocks(128 * 192), 256, 0, stream>>>(W1, w1b, 128, 168, 192);
  pad_convert_bf16<<<blocks(128 * 128), 256, 0, stream>>>(W2, w2b, 128, 128, 128);
  pad_convert_bf16<<<blocks(128), 256, 0, stream>>>(W3, w3b, 1, 128, 128);

  // grid for WMMA GEMM: (M/16)*(N/64) waves, 8 waves per block
  const unsigned g512 = (unsigned)((M / 16) * (512 / 64) / 8);  // 8192
  const unsigned g128 = (unsigned)((M / 16) * (128 / 64) / 8);  // 2048

  // --- layer 0 ---
  gemm_wmma_bf16<64,  false, false, true><<<g512, 256, 0, stream>>>(xbf, wihb0, biasc[0], gbuf, (int)M, 512);
  lstm_layer_kernel<<<16, 256, 0, stream>>>(gbuf, whhb[0], yA, Tn);
  // --- layer 1 ---
  gemm_wmma_bf16<128, false, false, true><<<g512, 256, 0, stream>>>(yA, wihb1, biasc[1], gbuf, (int)M, 512);
  lstm_layer_kernel<<<16, 256, 0, stream>>>(gbuf, whhb[1], yB, Tn);
  // --- layer 2 ---
  gemm_wmma_bf16<128, false, false, true><<<g512, 256, 0, stream>>>(yB, wihb2, biasc[2], gbuf, (int)M, 512);
  lstm_layer_kernel<<<16, 256, 0, stream>>>(gbuf, whhb[2], yA, Tn);

  // --- MLP head ---
  concat_kernel<<<blocks(M * 192), 256, 0, stream>>>(x, yA, mlpin, M);
  gemm_wmma_bf16<192, true, true, false><<<g128, 256, 0, stream>>>(mlpin, w1b, b1, h1, (int)M, 128);
  gemm_wmma_bf16<128, true, true, false><<<g128, 256, 0, stream>>>(h1, w2b, b2, h2, (int)M, 128);
  final_proj<<<blocks(M), 256, 0, stream>>>(h2, w3b, b3, (float*)d_out, M);
}
